// MultiHeadAttention_56083682951609
// MI455X (gfx1250) — matrix operand order repaired and checked by reading, not compiled, not run on hardware
//
#include <hip/hip_runtime.h>
#include <cstdint>
#include <cstddef>

// ---------------------------------------------------------------------------
// Types
// ---------------------------------------------------------------------------
typedef __attribute__((ext_vector_type(16))) _Float16 v16h;
typedef __attribute__((ext_vector_type(8)))  float    v8f;

struct alignas(16) H8 { _Float16 h[8]; };   // 16-byte chunk of halves
struct alignas(16) F4 { float    f[4]; };   // 16-byte chunk of floats
struct alignas(8)  H4 { _Float16 h[4]; };   // 8-byte chunk of halves

// Problem constants
constexpr int BB = 4, SS = 2048, EE = 1024, HH = 16, DH = 64;
constexpr int GM = BB * SS;   // 8192 rows for projection GEMMs
constexpr int GK = 1024;
constexpr int GN = 1024;

// 1/sqrt(DH) * log2(e): scores computed in base-2 domain for native v_exp_f32
constexpr float QSCALE = 0.18033688011112042f;

// ---------------------------------------------------------------------------
// WMMA helper (gfx1250: D = A(16x32 f16) * B(32x16 f16) + C(16x16 f32))
// ---------------------------------------------------------------------------
__device__ __forceinline__ v8f wmma32(v16h a, v16h b, v8f c) {
  return __builtin_amdgcn_wmma_f32_16x16x32_f16(false, a, false, b,
                                                (short)0, c, false, false);
}

// A-matrix fragment (16x32 f16), source row-major [row][k], stride in halves.
// ISA layout: lanes 0-15 row M=lane, K = {koff+0..7, koff+16..23};
//             lanes 16-31 row M=lane-16, K = {koff+8..15, koff+24..31}.
__device__ __forceinline__ v16h frag_a(const _Float16* p0, int row0,
                                       int stride_h, int koff, int lane) {
  union { v16h v; H8 p[2]; } u;
  const _Float16* p = p0 + (size_t)(row0 + (lane & 15)) * stride_h
                         + koff + ((lane >> 4) << 3);
  u.p[0] = *(const H8*)(p);
  u.p[1] = *(const H8*)(p + 16);
  return u.v;
}

// B-matrix fragment (32x16 f16), source is B^T row-major: [n][k].
// ISA layout: lanes 0-15 col N=lane, K = {koff+0..7, koff+16..23};
//             lanes 16-31 col N=lane-16, K = {koff+8..15, koff+24..31}.
__device__ __forceinline__ v16h frag_b(const _Float16* p0, int row0,
                                       int stride_h, int koff, int lane) {
  union { v16h v; H8 p[2]; } u;
  const _Float16* p = p0 + (size_t)(row0 + (lane & 15)) * stride_h
                         + koff + ((lane >> 4) << 3);
  u.p[0] = *(const H8*)(p);
  u.p[1] = *(const H8*)(p + 16);
  return u.v;
}

// ---------------------------------------------------------------------------
// CDNA5 async global->LDS copy (bypasses VGPRs, tracked by ASYNCcnt).
// Copies 32 bytes: the instruction adds INST_OFFSET to both the LDS and the
// global address, so one address pair serves both halves.
// ---------------------------------------------------------------------------
__device__ __forceinline__ void async_copy32(_Float16* lds, const _Float16* g) {
  uint32_t l = (uint32_t)(uintptr_t)lds;   // low 32 bits of flat LDS address
  asm volatile("global_load_async_to_lds_b128 %0, %1, off\n\t"
               "global_load_async_to_lds_b128 %0, %1, off offset:16"
               :
               : "v"(l), "v"(g)
               : "memory");
}
__device__ __forceinline__ void async_wait0() {
  asm volatile("s_wait_asynccnt 0" ::: "memory");
}

// ---------------------------------------------------------------------------
// DPP butterfly reductions within 16-lane halves (pure VALU, no LDS pipe).
// ctrl: 0xB1 quad_perm(1,0,3,2), 0x4E quad_perm(2,3,0,1),
//       0x141 ROW_HALF_MIRROR, 0x140 ROW_MIRROR.
// ---------------------------------------------------------------------------
template <int CTRL>
__device__ __forceinline__ float dpp_mov(float x) {
  return __int_as_float(__builtin_amdgcn_update_dpp(
      0, __float_as_int(x), CTRL, 0xF, 0xF, true));
}
__device__ __forceinline__ float red_max16(float x) {
  x = fmaxf(x, dpp_mov<0xB1>(x));
  x = fmaxf(x, dpp_mov<0x4E>(x));
  x = fmaxf(x, dpp_mov<0x141>(x));
  x = fmaxf(x, dpp_mov<0x140>(x));
  return x;
}
__device__ __forceinline__ float red_sum16(float x) {
  x += dpp_mov<0xB1>(x);
  x += dpp_mov<0x4E>(x);
  x += dpp_mov<0x141>(x);
  x += dpp_mov<0x140>(x);
  return x;
}

// ---------------------------------------------------------------------------
// fp32 -> f16 convert (vectorized)
// ---------------------------------------------------------------------------
__global__ __launch_bounds__(256) void cvt_kernel(const float* __restrict__ x,
                                                  _Float16* __restrict__ y) {
  int i = (blockIdx.x * 256 + threadIdx.x) * 4;
  F4 v = *(const F4*)(x + i);
  H4 o;
  o.h[0] = (_Float16)v.f[0];
  o.h[1] = (_Float16)v.f[1];
  o.h[2] = (_Float16)v.f[2];
  o.h[3] = (_Float16)v.f[3];
  *(H4*)(y + i) = o;
}

// fp32 W[K][N] -> f16 Wt[N][K]
__global__ __launch_bounds__(256) void wtr_kernel(const float* __restrict__ W,
                                                  _Float16* __restrict__ Wt) {
  int id = blockIdx.x * 256 + threadIdx.x;   // over 1024*1024
  int k = id >> 10, n = id & 1023;
  Wt[(size_t)n * 1024 + k] = (_Float16)W[id];
}

// ---------------------------------------------------------------------------
// WMMA GEMM: C[M=8192,N=1024] = A[M,K] * Bt[N,K]^T + bias
// Block 256 threads = 8 waves; block tile 128x128, wave tile 64x32.
// Double-buffered LDS fed by async global->LDS DMA.
// MODE 0: f16 out -> Qh [B,H,S,DH], scaled by QSCALE (1/sqrt(DH) * log2 e)
// MODE 1: f16 out -> Kh [B,H,S,DH]
// MODE 2: f16 out -> Vt [B,H,DH,S]
// MODE 3: f32 out -> row-major [M,N] (final output)
// ---------------------------------------------------------------------------
template <int MODE>
__global__ __launch_bounds__(256)
void gemm128(const _Float16* __restrict__ A, const _Float16* __restrict__ Bt,
             const float* __restrict__ bias, void* __restrict__ Cout) {
  __shared__ _Float16 As[2][128 * 40];
  __shared__ _Float16 Bs[2][128 * 40];

  const int tid  = threadIdx.x;
  const int lane = tid & 31;
  const int wave = tid >> 5;
  const int wm = wave >> 2;        // 0..1
  const int wn = wave & 3;         // 0..3
  const int m0 = blockIdx.x * 128;
  const int n0 = blockIdx.y * 128;

  const v8f vzero = {0.f, 0.f, 0.f, 0.f, 0.f, 0.f, 0.f, 0.f};
  v8f acc[4][2];
#pragma unroll
  for (int mi = 0; mi < 4; ++mi)
#pragma unroll
    for (int ni = 0; ni < 2; ++ni) acc[mi][ni] = vzero;

  const int lr = tid >> 1;            // 0..127 : tile row
  const int lc = (tid & 1) << 4;      // 0 or 16 : half-column offset
  const _Float16* ga = A  + (size_t)(m0 + lr) * GK + lc;
  const _Float16* gb = Bt + (size_t)(n0 + lr) * GK + lc;
  const int loff = lr * 40 + lc;

  // preload first tile into buffer 0
  async_copy32(&As[0][loff], ga);
  async_copy32(&Bs[0][loff], gb);
  async_wait0();
  __syncthreads();

  int buf = 0;
  for (int ko = 0; ko < GK; ko += 32, buf ^= 1) {
    if (ko + 32 < GK) {   // prefetch next tile while computing this one
      async_copy32(&As[buf ^ 1][loff], ga + ko + 32);
      async_copy32(&Bs[buf ^ 1][loff], gb + ko + 32);
    }

    v16h fa[4], fb[2];
#pragma unroll
    for (int mi = 0; mi < 4; ++mi)
      fa[mi] = frag_a(As[buf], wm * 64 + mi * 16, 40, 0, lane);
#pragma unroll
    for (int ni = 0; ni < 2; ++ni)
      fb[ni] = frag_b(Bs[buf], wn * 32 + ni * 16, 40, 0, lane);
#pragma unroll
    for (int mi = 0; mi < 4; ++mi)
#pragma unroll
      for (int ni = 0; ni < 2; ++ni)
        acc[mi][ni] = wmma32(fa[mi], fb[ni], acc[mi][ni]);

    async_wait0();
    __syncthreads();
  }

  // Epilogue. C layout: lanes 0-15 vgpr i -> M=i, lanes 16-31 vgpr i -> M=8+i,
  // N = lane&15.
  const int mrow = (lane >> 4) << 3;
  const int ncol = lane & 15;
#pragma unroll
  for (int ni = 0; ni < 2; ++ni) {
    const int gn = n0 + wn * 32 + ni * 16 + ncol;
    const float bv = bias[gn];
#pragma unroll
    for (int mi = 0; mi < 4; ++mi) {
#pragma unroll
      for (int i = 0; i < 8; ++i) {
        const int gm = m0 + wm * 64 + mi * 16 + mrow + i;
        float val = acc[mi][ni][i] + bv;
        if (MODE == 0 || MODE == 1) {
          if (MODE == 0) val *= QSCALE;
          const int b = gm >> 11, s = gm & 2047;
          const int hd = gn >> 6, d = gn & 63;
          ((_Float16*)Cout)[(((size_t)(b * HH + hd)) * SS + s) * DH + d] =
              (_Float16)val;
        } else if (MODE == 2) {
          const int b = gm >> 11, s = gm & 2047;
          const int hd = gn >> 6, d = gn & 63;
          ((_Float16*)Cout)[(((size_t)(b * HH + hd)) * DH + d) * SS + s] =
              (_Float16)val;
        } else {
          ((float*)Cout)[(size_t)gm * GN + gn] = val;
        }
      }
    }
  }
}

// ---------------------------------------------------------------------------
// Flash-style attention. Grid: x = S/128, y = B*H. Block = 256 = 8 waves.
// Each wave owns 16 query rows; key loop streams 64-key tiles through
// double-buffered LDS via async DMA.
// Qh, Kh: [B,H,S,DH] f16 (Q pre-scaled to base-2 domain). Vt: [B,H,DH,S] f16.
// Ah out: [B,S,H*DH] f16 (A-matrix for the output projection GEMM).
// ---------------------------------------------------------------------------
__global__ __launch_bounds__(256)
void attn_kernel(const _Float16* __restrict__ Qh, const _Float16* __restrict__ Kh,
                 const _Float16* __restrict__ Vt, _Float16* __restrict__ Ah) {
  __shared__ _Float16 Ks[2][64 * 64];     // [key][dh]
  __shared__ _Float16 Vs[2][64 * 64];     // [dh][key]
  __shared__ _Float16 Ps[8 * 16 * 64];    // per-wave P staging [16][64]

  const int tid  = threadIdx.x;
  const int lane = tid & 31;
  const int wave = tid >> 5;
  const int bh = blockIdx.y;              // 0..63
  const int q0 = blockIdx.x * 128;
  const size_t baseQK = (size_t)bh * SS * DH;
  const size_t baseV  = (size_t)bh * DH * SS;

  // Q fragments for this wave's 16 rows (kept in registers for all key tiles)
  const int qrow = q0 + wave * 16;
  const _Float16* qp = Qh + baseQK + (size_t)qrow * DH;
  v16h fq[2];
  fq[0] = frag_a(qp, 0, DH, 0, lane);
  fq[1] = frag_a(qp, 0, DH, 32, lane);

  const v8f vzero = {0.f, 0.f, 0.f, 0.f, 0.f, 0.f, 0.f, 0.f};
  v8f o[4];
#pragma unroll
  for (int n = 0; n < 4; ++n) o[n] = vzero;
  float m[8], l[8];
#pragma unroll
  for (int i = 0; i < 8; ++i) { m[i] = -3.0e38f; l[i] = 0.f; }

  _Float16* Pw = Ps + wave * (16 * 64);

  const int trow = tid >> 2;              // 0..63
  const int tcol = (tid & 3) << 4;        // 0,16,32,48
  const _Float16* gk = Kh + baseQK + (size_t)trow * DH + tcol;   // key row trow
  const _Float16* gv = Vt + baseV + (size_t)trow * SS + tcol;    // dh row trow
  const int loff = trow * 64 + tcol;

  // preload first K/V tile into buffer 0
  async_copy32(&Ks[0][loff], gk);
  async_copy32(&Vs[0][loff], gv);
  async_wait0();
  __syncthreads();

  int buf = 0;
  for (int kt = 0; kt < SS; kt += 64, buf ^= 1) {
    if (kt + 64 < SS) {   // prefetch next K/V tile during compute
      async_copy32(&Ks[buf ^ 1][loff], gk + (size_t)(kt + 64) * DH);
      async_copy32(&Vs[buf ^ 1][loff], gv + kt + 64);
    }

    // scores: 16 q rows x 64 keys (4 C-tiles), K-dim = DH = 64
    v8f sc[4];
#pragma unroll
    for (int n = 0; n < 4; ++n) {
      v8f c = vzero;
      c = wmma32(fq[0], frag_b(Ks[buf], n * 16, 64, 0, lane), c);
      c = wmma32(fq[1], frag_b(Ks[buf], n * 16, 64, 32, lane), c);
      sc[n] = c;
    }

    // online softmax per row-slot; row reductions = DPP butterflies (VALU only)
#pragma unroll
    for (int i = 0; i < 8; ++i) {
      float t = fmaxf(fmaxf(sc[0][i], sc[1][i]), fmaxf(sc[2][i], sc[3][i]));
      t = red_max16(t);
      const float mn = fmaxf(m[i], t);
      const float alpha = exp2f(m[i] - mn);
      float p[4];
      float rs = 0.f;
#pragma unroll
      for (int n = 0; n < 4; ++n) { p[n] = exp2f(sc[n][i] - mn); rs += p[n]; }
      rs = red_sum16(rs);
      l[i] = l[i] * alpha + rs;
      m[i] = mn;
#pragma unroll
      for (int n2 = 0; n2 < 4; ++n2) o[n2][i] *= alpha;
      // stage P (C-layout -> canonical row-major f16 in LDS)
      const int pr = i + ((lane >> 4) << 3);
#pragma unroll
      for (int n = 0; n < 4; ++n)
        Pw[pr * 64 + n * 16 + (lane & 15)] = (_Float16)p[n];
    }
    asm volatile("s_wait_dscnt 0" ::: "memory");

    // P(16x64) @ V(64x64): A-frags from staged P, B-frags from Vt tile
    const v16h fp0 = frag_a(Pw, 0, 64, 0, lane);
    const v16h fp1 = frag_a(Pw, 0, 64, 32, lane);
#pragma unroll
    for (int n2 = 0; n2 < 4; ++n2) {
      o[n2] = wmma32(fp0, frag_b(Vs[buf], n2 * 16, 64, 0, lane), o[n2]);
      o[n2] = wmma32(fp1, frag_b(Vs[buf], n2 * 16, 64, 32, lane), o[n2]);
    }

    async_wait0();
    __syncthreads();
  }

  // normalize and scatter to Ah [B,S,H*DH]
  const int b = bh >> 4, hd = bh & 15;
#pragma unroll
  for (int i = 0; i < 8; ++i) {
    const float inv = 1.0f / l[i];
    const int srow = qrow + i + ((lane >> 4) << 3);
    const size_t rb = ((size_t)(b * SS + srow)) * (HH * DH) + hd * DH;
#pragma unroll
    for (int n2 = 0; n2 < 4; ++n2)
      Ah[rb + n2 * 16 + (lane & 15)] = (_Float16)(o[n2][i] * inv);
  }
}

// ---------------------------------------------------------------------------
// Launcher
// ---------------------------------------------------------------------------
extern "C" void kernel_launch(void* const* d_in, const int* in_sizes, int n_in,
                              void* d_out, int out_size, void* d_ws, size_t ws_size,
                              hipStream_t stream) {
  const float* q  = (const float*)d_in[0];
  const float* k  = (const float*)d_in[1];
  const float* v  = (const float*)d_in[2];
  const float* Wq = (const float*)d_in[3];
  const float* bq = (const float*)d_in[4];
  const float* Wk = (const float*)d_in[5];
  const float* bk = (const float*)d_in[6];
  const float* Wv = (const float*)d_in[7];
  const float* bv = (const float*)d_in[8];
  const float* Wo = (const float*)d_in[9];
  const float* bo = (const float*)d_in[10];

  const size_t NA = (size_t)BB * SS * EE;   // 8388608 activations
  const size_t NW = (size_t)GK * GN;        // 1048576 per weight

  _Float16* ws  = (_Float16*)d_ws;
  _Float16* qf  = ws;
  _Float16* kf  = qf + NA;
  _Float16* vf  = kf + NA;
  _Float16* wqT = vf + NA;
  _Float16* wkT = wqT + NW;
  _Float16* wvT = wkT + NW;
  _Float16* woT = wvT + NW;
  _Float16* qh  = woT + NW;
  _Float16* kh  = qh + NA;
  _Float16* vt  = kh + NA;
  _Float16* ah  = vt + NA;

  const int cvtBlocks = (int)(NA / 4 / 256);
  cvt_kernel<<<cvtBlocks, 256, 0, stream>>>(q, qf);
  cvt_kernel<<<cvtBlocks, 256, 0, stream>>>(k, kf);
  cvt_kernel<<<cvtBlocks, 256, 0, stream>>>(v, vf);

  const int wtrBlocks = (int)(NW / 256);
  wtr_kernel<<<wtrBlocks, 256, 0, stream>>>(Wq, wqT);
  wtr_kernel<<<wtrBlocks, 256, 0, stream>>>(Wk, wkT);
  wtr_kernel<<<wtrBlocks, 256, 0, stream>>>(Wv, wvT);
  wtr_kernel<<<wtrBlocks, 256, 0, stream>>>(Wo, woT);

  dim3 gg(GM / 128, GN / 128);
  gemm128<0><<<gg, 256, 0, stream>>>(qf, wqT, bq, (void*)qh);
  gemm128<1><<<gg, 256, 0, stream>>>(kf, wkT, bk, (void*)kh);
  gemm128<2><<<gg, 256, 0, stream>>>(vf, wvT, bv, (void*)vt);

  dim3 ga(SS / 128, BB * HH);
  attn_kernel<<<ga, 256, 0, stream>>>(qh, kh, vt, ah);

  gemm128<3><<<gg, 256, 0, stream>>>(ah, woT, bo, d_out);
}
